// PMSAViT_38749194945190
// MI455X (gfx1250) — matrix operand order repaired and checked by reading, not compiled, not run on hardware
//
#include <hip/hip_runtime.h>
#include <hip/hip_bf16.h>
#include <math.h>
#include <stdint.h>

// ============================================================================
// PMSAViT forward for MI455X (gfx1250, wave32, WMMA + TDM).
//
// Precision plan: fp32 residual stream / stats; all GEMM inner products via
// v_wmma_f32_16x16x32_f16 (f16 in, fp32 accumulate). Weights converted +
// transposed to f16 once per launch. GEMM tiles are staged into LDS by the
// Tensor Data Mover (tensor_load_to_lds, TENSORcnt-tracked, double-buffered
// so DMA of tile k+1 overlaps WMMA on tile k). Attention is strip-fused in
// LDS with a two-pass global-min to honor the reference's dots.min() fill.
//
// Workspace requirement: ~190 MB (all intermediate tensors + f16 weights).
// ============================================================================

#define DEV_INLINE __device__ __forceinline__

#if defined(__has_builtin)
#if __has_builtin(__builtin_amdgcn_tensor_load_to_lds)
#define VIT_HAVE_TDM 1
#endif
#endif
#ifndef VIT_HAVE_TDM
#define VIT_HAVE_TDM 0
#endif

namespace vit {

constexpr int BATCH  = 8;
constexpr int C_IN   = 320;
constexpr int IMG    = 64;
constexpr int HH     = 32;          // H = W = 32
constexpr int NTOK   = 1024;        // tokens per image
constexpr int DIM    = 512;
constexpr int DEPTH  = 6;
constexpr int HEADS  = 8;
constexpr int DH     = 64;
constexpr int INNER  = 512;
constexpr int MLP    = 512;
constexpr int MROWS  = BATCH * NTOK;   // 8192 GEMM rows

typedef __attribute__((ext_vector_type(16))) _Float16 v16h;
typedef __attribute__((ext_vector_type(8)))  float    v8f;
typedef unsigned int u32;
typedef __attribute__((ext_vector_type(4))) u32 u32x4;
typedef __attribute__((ext_vector_type(4))) int i32x4;
typedef __attribute__((ext_vector_type(8))) int i32x8;

// ---------------------------------------------------------------------------
// WMMA fragment loaders (layouts per CDNA5 ISA 7.12.2, wave32).
// A 16x32 f16: lanes 0-15 -> row M=l, K {0..7}u{16..23}; lanes 16-31 ->
//              row M=l-16, K {8..15}u{24..31}.  (two 16B chunks)
// B 32x16 f16: lanes 0-15 -> col N=l, K {0..7}u{16..23}; lanes 16-31 ->
//              col N=l-16, K {8..15}u{24..31}.  (two 16B chunks)
// C/D 16x16 f32: lane l -> col N=l&15, VGPR v -> row M = v + (l>=16 ? 8 : 0).
// ---------------------------------------------------------------------------
DEV_INLINE v16h load_frag_a(const _Float16* base, int ld, int lane) {
  int r  = lane & 15;
  int ko = (lane & 16) ? 8 : 0;
  const _Float16* p = base + r * ld + ko;
  union { float4 f[2]; v16h h; } u;
  u.f[0] = *reinterpret_cast<const float4*>(p);
  u.f[1] = *reinterpret_cast<const float4*>(p + 16);
  return u.h;
}

DEV_INLINE v16h load_frag_b(const _Float16* base, int ld, int lane) {
  int n  = lane & 15;
  int ko = (lane & 16) ? 8 : 0;
  const _Float16* p = base + n * ld + ko;
  union { float4 f[2]; v16h h; } u;
  u.f[0] = *reinterpret_cast<const float4*>(p);
  u.f[1] = *reinterpret_cast<const float4*>(p + 16);
  return u.h;
}

DEV_INLINE float gelu_exact(float v) {
  return 0.5f * v * (1.0f + erff(v * 0.70710678118654752f));
}

#if VIT_HAVE_TDM
// ---------------------------------------------------------------------------
// TDM: DMA a [rows x 32-halves] f16 tile (row stride = strideElems halves in
// global) into LDS at ldsAddr, inserting 4 DWORDs of pad after every 16
// stored DWORDs -> LDS row stride of 40 halves (matches frag loaders).
// D# layout per CDNA5 ISA; tile==tensor view so
// no OOB. Issued by one wave; tracked with TENSORcnt.
// ---------------------------------------------------------------------------
DEV_INLINE void tdm_load_tile_f16(u32 ldsAddr, const _Float16* gptr,
                                  u32 rows, u32 strideElems) {
  unsigned long long ga = (unsigned long long)(uintptr_t)gptr;
  u32x4 g0;
  g0[0] = 1u;                                   // count=1 (valid user D#)
  g0[1] = ldsAddr;                              // LDS byte address
  g0[2] = (u32)(ga & 0xffffffffu);              // global_addr[31:0]
  g0[3] = (u32)((ga >> 32) & 0x01ffffffu)       // global_addr[56:32]
        | (2u << 30);                           // type = 2 ("image")
  i32x8 g1;
  g1[0] = (int)((1u << 16)                      // data_size = 2 bytes
              | (1u << 20)                      // pad_enable
              | (3u << 22)                      // pad_interval: 16 DWORDs
              | (3u << 25));                    // pad_amount:   4 DWORDs
  g1[1] = (int)((32u & 0xffffu) << 16);         // tensor_dim0 = 32 (lo16)
  g1[2] = (int)((rows & 0xffffu) << 16);        // dim0 hi=0 | tensor_dim1 lo
  g1[3] = (int)(((rows >> 16) & 0xffffu)        // tensor_dim1 hi
              | (32u << 16));                   // tile_dim0 = 32
  g1[4] = (int)(rows & 0xffffu);                // tile_dim1 = rows, tile_dim2=0
  g1[5] = (int)strideElems;                     // tensor_dim0_stride[31:0]
  g1[6] = 0;                                    // stride hi / dim1_stride lo
  g1[7] = 0;
  i32x4 z4 = {0, 0, 0, 0};
#if __clang_major__ >= 23
  i32x8 z8 = {0, 0, 0, 0, 0, 0, 0, 0};
  __builtin_amdgcn_tensor_load_to_lds(g0, g1, z4, z4, z8, 0);
#else
  __builtin_amdgcn_tensor_load_to_lds(g0, g1, z4, z4, 0);
#endif
}
#endif  // VIT_HAVE_TDM

// ---------------------------------------------------------------------------
// Generic WMMA GEMM: out[M,Nc] = A[M,K] (f16, row-major) x Bt[Nc,K] (f16,
// weight pre-transposed) with optional bias / residual / gelu epilogue.
// Block: 256 threads (8 waves), tile 128x64, BK=32; wave -> 32x32 patch.
// Tiles staged by TDM, double-buffered: DMA(k+1) overlaps WMMA(k).
// Requires M%128==0, Nc%64==0, K%32==0 (true for all layers here).
// ---------------------------------------------------------------------------
template <bool BIAS, bool RES, bool GELU, bool O32, bool O16>
__global__ void __launch_bounds__(256)
gemm_f16_wmma(const _Float16* __restrict__ A, const _Float16* __restrict__ Bt,
              const float* __restrict__ bias, const float* __restrict__ res,
              float* __restrict__ out32, _Float16* __restrict__ out16,
              int M, int Nc, int K) {
  constexpr int BM = 128, BN = 64, BK = 32, LDT = BK + 8;  // 40-half rows
  __shared__ __align__(16) _Float16 As[2][BM * LDT];
  __shared__ __align__(16) _Float16 Bs[2][BN * LDT];

  const int tid  = threadIdx.x;
  const int lane = tid & 31;
  const int wave = tid >> 5;
  const int wm   = wave >> 1;   // 0..3
  const int wn   = wave & 1;    // 0..1
  const int rowBase = blockIdx.y * BM;
  const int colBase = blockIdx.x * BN;
  const _Float16* aTile = A + (size_t)rowBase * K;
  const _Float16* bTile = Bt + (size_t)colBase * K;

  v8f acc[4];
  for (int t = 0; t < 4; ++t)
    for (int j = 0; j < 8; ++j) acc[t][j] = 0.0f;

#if VIT_HAVE_TDM
  const u32 asAddr0 = (u32)(uintptr_t)(&As[0][0]);
  const u32 asAddr1 = (u32)(uintptr_t)(&As[1][0]);
  const u32 bsAddr0 = (u32)(uintptr_t)(&Bs[0][0]);
  const u32 bsAddr1 = (u32)(uintptr_t)(&Bs[1][0]);
  if (wave == 0) {                      // prefetch tile 0 into buffer 0
    tdm_load_tile_f16(asAddr0, aTile, BM, (u32)K);
    tdm_load_tile_f16(bsAddr0, bTile, BN, (u32)K);
  }
  int cur = 0;
  for (int k0 = 0; k0 < K; k0 += BK) {
    const bool more = (k0 + BK) < K;
    if (wave == 0) {
      if (more) {                       // kick DMA for tile k+1 (other buffer)
        tdm_load_tile_f16(cur ? asAddr0 : asAddr1, aTile + k0 + BK, BM, (u32)K);
        tdm_load_tile_f16(cur ? bsAddr0 : bsAddr1, bTile + k0 + BK, BN, (u32)K);
        __builtin_amdgcn_s_wait_tensorcnt(2);   // current pair done (in-order)
      } else {
        __builtin_amdgcn_s_wait_tensorcnt(0);
      }
    }
    __syncthreads();
    const _Float16* as = As[cur];
    const _Float16* bs = Bs[cur];
    v16h a0 = load_frag_a(&as[(wm * 32) * LDT],      LDT, lane);
    v16h a1 = load_frag_a(&as[(wm * 32 + 16) * LDT], LDT, lane);
    v16h b0 = load_frag_b(&bs[(wn * 32) * LDT],      LDT, lane);
    v16h b1 = load_frag_b(&bs[(wn * 32 + 16) * LDT], LDT, lane);
    acc[0] = __builtin_amdgcn_wmma_f32_16x16x32_f16(false, a0, false, b0, (short)0, acc[0], false, false);
    acc[1] = __builtin_amdgcn_wmma_f32_16x16x32_f16(false, a0, false, b1, (short)0, acc[1], false, false);
    acc[2] = __builtin_amdgcn_wmma_f32_16x16x32_f16(false, a1, false, b0, (short)0, acc[2], false, false);
    acc[3] = __builtin_amdgcn_wmma_f32_16x16x32_f16(false, a1, false, b1, (short)0, acc[3], false, false);
    __syncthreads();
    cur ^= 1;
  }
#else
  // Fallback: per-thread vectorized staging, strength-reduced addressing.
  const int arow = tid >> 2;                 // 0..63 (chunk rows; +64 second)
  const int akc  = (tid & 3) << 3;           // 0,8,16,24
  const _Float16* aPtr = aTile + (size_t)arow * K + akc;
  const _Float16* bPtr = bTile + (size_t)arow * K + akc;  // rows 0..63
  _Float16* aDst0 = &As[0][arow * LDT + akc];
  _Float16* aDst1 = &As[0][(arow + 64) * LDT + akc];
  _Float16* bDst  = &Bs[0][arow * LDT + akc];
  const size_t aSkip = (size_t)64 * K;
  for (int k0 = 0; k0 < K; k0 += BK) {
    *reinterpret_cast<float4*>(aDst0) = *reinterpret_cast<const float4*>(aPtr);
    *reinterpret_cast<float4*>(aDst1) = *reinterpret_cast<const float4*>(aPtr + aSkip);
    *reinterpret_cast<float4*>(bDst)  = *reinterpret_cast<const float4*>(bPtr);
    aPtr += BK; bPtr += BK;
    __syncthreads();
    const _Float16* as = As[0];
    const _Float16* bs = Bs[0];
    v16h a0 = load_frag_a(&as[(wm * 32) * LDT],      LDT, lane);
    v16h a1 = load_frag_a(&as[(wm * 32 + 16) * LDT], LDT, lane);
    v16h b0 = load_frag_b(&bs[(wn * 32) * LDT],      LDT, lane);
    v16h b1 = load_frag_b(&bs[(wn * 32 + 16) * LDT], LDT, lane);
    acc[0] = __builtin_amdgcn_wmma_f32_16x16x32_f16(false, a0, false, b0, (short)0, acc[0], false, false);
    acc[1] = __builtin_amdgcn_wmma_f32_16x16x32_f16(false, a0, false, b1, (short)0, acc[1], false, false);
    acc[2] = __builtin_amdgcn_wmma_f32_16x16x32_f16(false, a1, false, b0, (short)0, acc[2], false, false);
    acc[3] = __builtin_amdgcn_wmma_f32_16x16x32_f16(false, a1, false, b1, (short)0, acc[3], false, false);
    __syncthreads();
  }
#endif

  const int ro = (lane & 16) ? 8 : 0;
  for (int t = 0; t < 4; ++t) {
    int mb  = rowBase + wm * 32 + (t >> 1) * 16;
    int col = colBase + wn * 32 + (t & 1) * 16 + (lane & 15);
    for (int v = 0; v < 8; ++v) {
      int row = mb + ro + v;
      float val = acc[t][v];
      if (BIAS) val += bias[col];
      if (RES)  val += res[(size_t)row * Nc + col];
      if (GELU) val = gelu_exact(val);
      size_t o = (size_t)row * Nc + col;
      if (O32) out32[o] = val;
      if (O16) out16[o] = (_Float16)val;
    }
  }
}

// ---------------------------------------------------------------------------
// Attention, strip-fused. One block per (16-token strip, head, batch).
// MODE 0: recompute scores with WMMA, block-reduce global raw-score min.
// MODE 1: scores -> LDS fp32 strip, mask+softmax, f16 probs -> LDS, P@V WMMA.
// scale (DH^-0.5) is folded into q during split_qkv.
// ---------------------------------------------------------------------------
template <int MODE>
__global__ void __launch_bounds__(256)
attn_wmma(const _Float16* __restrict__ q, const _Float16* __restrict__ k,
          const _Float16* __restrict__ vT, const int* __restrict__ rm_layer,
          const float* __restrict__ minval, float* __restrict__ minPartials,
          _Float16* __restrict__ o16) {
  constexpr int SR = 16;         // strip rows
  constexpr int LQ = 72;         // Qs stride (halves), 144B (16B-aligned rows)
  constexpr int LP = 1032;       // Ps stride (halves), 2064B
  extern __shared__ __align__(16) char smem[];
  _Float16* Qs  = (_Float16*)(smem);                    //  2304 B
  float*    red = (float*)(smem + 2304);                //  1024 B
  int*      cm  = (int*)(smem + 3328);                  //  4096 B
  float*    Ss  = (float*)(smem + 7424);                // 65536 B (16x1024 f32)
  _Float16* Ps  = (_Float16*)(smem + 72960);            // 33024 B (16x1032 f16)

  const int tid  = threadIdx.x;
  const int lane = tid & 31;
  const int wave = tid >> 5;
  const int strip = blockIdx.x, h = blockIdx.y, b = blockIdx.z;
  const int row0 = strip * SR;
  const size_t bh = (size_t)(b * HEADS + h);
  const _Float16* qbh = q + bh * NTOK * DH;
  const _Float16* kbh = k + bh * NTOK * DH;

  // Stage Q strip (16x64 halves).
  if (tid < 128) {
    int r = tid >> 3, kc = (tid & 7) << 3;
    *reinterpret_cast<float4*>(&Qs[r * LQ + kc]) =
        *reinterpret_cast<const float4*>(&qbh[(size_t)(row0 + r) * DH + kc]);
  }
  if (MODE == 1) {
    const int* rmb = rm_layer + b * NTOK;
    for (int j = tid; j < NTOK; j += 256) cm[j] = rmb[j];
  }
  __syncthreads();

  v16h aq0 = load_frag_a(&Qs[0],  LQ, lane);   // dh 0..31
  v16h aq1 = load_frag_a(&Qs[32], LQ, lane);   // dh 32..63

  float lmin = 3.0e38f;
  // S strip: each wave covers 128 columns (8 n-tiles of 16).
  for (int nt = 0; nt < 8; ++nt) {
    int col0 = wave * 128 + nt * 16;
    const _Float16* kb = kbh + (size_t)col0 * DH;
    v16h b0 = load_frag_b(kb,      DH, lane);
    v16h b1 = load_frag_b(kb + 32, DH, lane);
    v8f s;
    for (int j = 0; j < 8; ++j) s[j] = 0.0f;
    s = __builtin_amdgcn_wmma_f32_16x16x32_f16(false, aq0, false, b0, (short)0, s, false, false);
    s = __builtin_amdgcn_wmma_f32_16x16x32_f16(false, aq1, false, b1, (short)0, s, false, false);
    if (MODE == 0) {
      for (int j = 0; j < 8; ++j) lmin = fminf(lmin, s[j]);
    } else {
      int col = col0 + (lane & 15);
      int ro  = (lane & 16) ? 8 : 0;
      for (int v = 0; v < 8; ++v) Ss[(ro + v) * NTOK + col] = s[v];
    }
  }

  if (MODE == 0) {
    red[tid] = lmin;
    __syncthreads();
    for (int t = 128; t > 0; t >>= 1) {
      if (tid < t) red[tid] = fminf(red[tid], red[tid + t]);
      __syncthreads();
    }
    if (tid == 0) {
      int bid = (blockIdx.z * gridDim.y + blockIdx.y) * gridDim.x + blockIdx.x;
      minPartials[bid] = red[0];
    }
    return;
  }
  __syncthreads();

  // Mask + softmax: 2 rows per wave.
  const float mv = minval[0];
  for (int rr = 0; rr < 2; ++rr) {
    int r = wave * 2 + rr;
    int tok = row0 + r;
    bool rowAllowed = cm[tok] != 0;
    float* srow = Ss + r * NTOK;
    float mx = -3.0e38f;
    for (int j = lane; j < NTOK; j += 32) {
      float sv = srow[j];
      if (!rowAllowed && cm[j] == 0) sv = mv;
      srow[j] = sv;
      mx = fmaxf(mx, sv);
    }
    for (int off = 16; off > 0; off >>= 1) mx = fmaxf(mx, __shfl_xor(mx, off, 32));
    float sum = 0.0f;
    for (int j = lane; j < NTOK; j += 32) {
      float e = __expf(srow[j] - mx);
      srow[j] = e;
      sum += e;
    }
    for (int off = 16; off > 0; off >>= 1) sum += __shfl_xor(sum, off, 32);
    float inv = 1.0f / sum;
    _Float16* prow = Ps + r * LP;
    for (int j = lane; j < NTOK; j += 32) prow[j] = (_Float16)(srow[j] * inv);
  }
  __syncthreads();

  // O(16x64) = P(16x1024) @ V(1024x64). 8 waves: 4 dh-tiles x 2 token halves.
  const _Float16* vbh = vT + bh * DH * NTOK;
  const int nt = wave & 3;      // dh tile
  const int kh = wave >> 2;     // token half
  v8f oacc;
  for (int j = 0; j < 8; ++j) oacc[j] = 0.0f;
  const _Float16* vbase = vbh + (size_t)(nt * 16) * NTOK;
  for (int ks = 0; ks < 16; ++ks) {
    int t0 = kh * 512 + ks * 32;
    v16h ap = load_frag_a(&Ps[t0],    LP,   lane);
    v16h bv = load_frag_b(vbase + t0, NTOK, lane);
    oacc = __builtin_amdgcn_wmma_f32_16x16x32_f16(false, ap, false, bv, (short)0, oacc, false, false);
  }
  // Combine token halves via LDS (reuse Ss), then write O as f16.
  if (kh == 1) {
    float* part = Ss + nt * 256;
    for (int v = 0; v < 8; ++v) part[lane * 8 + v] = oacc[v];
  }
  __syncthreads();
  if (kh == 0) {
    float* part = Ss + nt * 256;
    int d  = nt * 16 + (lane & 15);
    int ro = (lane & 16) ? 8 : 0;
    for (int v = 0; v < 8; ++v) {
      float val = oacc[v] + part[lane * 8 + v];
      int tok = row0 + ro + v;
      o16[((size_t)(b * NTOK) + tok) * INNER + h * DH + d] = (_Float16)val;
    }
  }
}

__global__ void reduce_min_k(const float* __restrict__ part, int n,
                             float* __restrict__ out) {
  __shared__ float red[256];
  float m = 3.0e38f;
  for (int j = threadIdx.x; j < n; j += 256) m = fminf(m, part[j]);
  red[threadIdx.x] = m;
  __syncthreads();
  for (int t = 128; t > 0; t >>= 1) {
    if (threadIdx.x < t) red[threadIdx.x] = fminf(red[threadIdx.x], red[threadIdx.x + t]);
    __syncthreads();
  }
  if (threadIdx.x == 0) out[0] = red[0];
}

// ---------------------------------------------------------------------------
// LayerNorm over last dim (D=512): one block per row. In-place safe.
// ---------------------------------------------------------------------------
template <bool O32, bool O16>
__global__ void __launch_bounds__(256)
ln_kernel(const float* __restrict__ in, const float* __restrict__ g,
          const float* __restrict__ b, float* __restrict__ out32,
          _Float16* __restrict__ out16, int D) {
  const int row = blockIdx.x, tid = threadIdx.x;
  const float* xr = in + (size_t)row * D;
  __shared__ float rs[256], rq[256];
  float s = 0.0f, s2 = 0.0f;
  for (int j = tid; j < D; j += 256) {
    float v = xr[j];
    s += v;
    s2 += v * v;
  }
  rs[tid] = s; rq[tid] = s2;
  __syncthreads();
  for (int t = 128; t > 0; t >>= 1) {
    if (tid < t) { rs[tid] += rs[tid + t]; rq[tid] += rq[tid + t]; }
    __syncthreads();
  }
  float mu  = rs[0] / D;
  float var = rq[0] / D - mu * mu;
  float inv = rsqrtf(var + 1e-5f);
  for (int j = tid; j < D; j += 256) {
    float v = (xr[j] - mu) * inv * g[j] + b[j];
    if (O32) out32[(size_t)row * D + j] = v;
    if (O16) out16[(size_t)row * D + j] = (_Float16)v;
  }
}

// Weight fp32 [DD][K][Nc] -> f16 transposed [DD][Nc][K].
__global__ void wcast_T(const float* __restrict__ src, _Float16* __restrict__ dst,
                        int DD, int K, int Nc) {
  size_t idx = (size_t)blockIdx.x * blockDim.x + threadIdx.x;
  size_t total = (size_t)DD * K * Nc;
  if (idx >= total) return;
  int d = (int)(idx / ((size_t)K * Nc));
  size_t r = idx % ((size_t)K * Nc);
  int kk = (int)(r / Nc), nn = (int)(r % Nc);
  dst[(size_t)d * K * Nc + (size_t)nn * K + kk] = (_Float16)src[idx];
}

// mask [DEPTH*B,64,64] int -> rmask [DEPTH*B, N] (OR over 2x2 block).
__global__ void build_rmask(const int* __restrict__ mask, int* __restrict__ rmask) {
  int idx = blockIdx.x * blockDim.x + threadIdx.x;
  if (idx >= DEPTH * BATCH * NTOK) return;
  int t = idx % NTOK, m = idx / NTOK;
  int hh = t / HH, ww = t % HH;
  const int* mm = mask + (size_t)m * 64 * 64;
  int a = 0;
  for (int dy = 0; dy < 2; ++dy)
    for (int dx = 0; dx < 2; ++dx)
      a |= (mm[(hh * 2 + dy) * 64 + (ww * 2 + dx)] == 1);
  rmask[idx] = a;
}

// qkv fp32 [B*N,3*INNER] -> q,k f16 [B,H,N,DH] (q pre-scaled), vT f16 [B,H,DH,N]
__global__ void split_qkv(const float* __restrict__ qkv, _Float16* __restrict__ q,
                          _Float16* __restrict__ k, _Float16* __restrict__ vT) {
  size_t idx = (size_t)blockIdx.x * blockDim.x + threadIdx.x;
  if (idx >= (size_t)BATCH * NTOK * INNER) return;
  int d = (int)(idx % DH);
  int h = (int)((idx / DH) % HEADS);
  int n = (int)((idx / INNER) % NTOK);
  int b = (int)(idx / ((size_t)INNER * NTOK));
  const float* row = qkv + ((size_t)(b * NTOK + n)) * (3 * INNER);
  float qv = row[h * DH + d];
  float kv = row[INNER + h * DH + d];
  float vv = row[2 * INNER + h * DH + d];
  size_t bh = (size_t)(b * HEADS + h);
  q[(bh * NTOK + n) * DH + d] = (_Float16)(qv * 0.125f);  // DH^-0.5 folded in
  k[(bh * NTOK + n) * DH + d] = (_Float16)kv;
  vT[(bh * DH + d) * NTOK + n] = (_Float16)vv;
}

// Direct fp32 conv3x3 SAME (small fraction of total FLOPs).
__global__ void conv3x3_k(const float* __restrict__ in, const float* __restrict__ w,
                          const float* __restrict__ bias, float* __restrict__ out,
                          int Bn, int Cin, int Cout) {
  size_t idx = (size_t)blockIdx.x * blockDim.x + threadIdx.x;
  size_t total = (size_t)Bn * Cout * IMG * IMG;
  if (idx >= total) return;
  int x  = (int)(idx % IMG);
  int y  = (int)((idx / IMG) % IMG);
  int co = (int)((idx / (IMG * IMG)) % Cout);
  int b  = (int)(idx / ((size_t)IMG * IMG * Cout));
  const float* wp = w + (size_t)co * Cin * 9;
  float acc = bias[co];
  for (int ci = 0; ci < Cin; ++ci) {
    const float* ip = in + ((size_t)(b * Cin + ci)) * (IMG * IMG);
    const float* wc = wp + ci * 9;
    for (int ky = 0; ky < 3; ++ky) {
      int yy = y + ky - 1;
      if (yy < 0 || yy > IMG - 1) continue;
      for (int kx = 0; kx < 3; ++kx) {
        int xx = x + kx - 1;
        if (xx < 0 || xx > IMG - 1) continue;
        acc += ip[yy * IMG + xx] * wc[ky * 3 + kx];
      }
    }
  }
  out[idx] = acc;
}

// conv out [B,128,64,64] -> y [B,N,512]  (b c (h p1) (w p2) -> b (h w) (p1 p2 c))
__global__ void patchify(const float* __restrict__ conv, float* __restrict__ y) {
  size_t idx = (size_t)blockIdx.x * blockDim.x + threadIdx.x;
  if (idx >= (size_t)BATCH * NTOK * DIM) return;
  int d = (int)(idx % DIM);
  int n = (int)((idx / DIM) % NTOK);
  int b = (int)(idx / ((size_t)DIM * NTOK));
  int c = d % 128, pp = d / 128, p1 = pp / 2, p2 = pp % 2;
  int h = n / HH, w = n % HH;
  y[idx] = conv[((size_t)(b * 128 + c)) * (IMG * IMG) + (h * 2 + p1) * IMG + (w * 2 + p2)];
}

// x [B,N,512] -> z [B,128,64,64]
__global__ void unpatchify(const float* __restrict__ x, float* __restrict__ z) {
  size_t idx = (size_t)blockIdx.x * blockDim.x + threadIdx.x;
  if (idx >= (size_t)BATCH * 128 * IMG * IMG) return;
  int px = (int)(idx % IMG);
  int py = (int)((idx / IMG) % IMG);
  int c  = (int)((idx / (IMG * IMG)) % 128);
  int b  = (int)(idx / ((size_t)IMG * IMG * 128));
  int h = py / 2, p1 = py % 2, w = px / 2, p2 = px % 2;
  z[idx] = x[((size_t)(b * NTOK + h * HH + w)) * DIM + (p1 * 2 + p2) * 128 + c];
}

// x += sincos 2d posemb (computed on the fly).
__global__ void posemb_add(float* __restrict__ x) {
  size_t idx = (size_t)blockIdx.x * blockDim.x + threadIdx.x;
  if (idx >= (size_t)BATCH * NTOK * DIM) return;
  int d = (int)(idx % DIM);
  int n = (int)((idx / DIM) % NTOK);
  int yy = n / HH, xx = n % HH;
  int quarter = d / 128, i = d % 128;
  float omega = __expf(-9.210340371976184f * (float)i / 127.0f);  // 10000^-(i/127)
  float pos = (quarter < 2) ? (float)xx : (float)yy;
  float a = pos * omega;
  float pe = ((quarter & 1) == 0) ? __sinf(a) : __cosf(a);
  x[idx] += pe;
}

}  // namespace vit

// ============================================================================
extern "C" void kernel_launch(void* const* d_in, const int* in_sizes, int n_in,
                              void* d_out, int out_size, void* d_ws, size_t ws_size,
                              hipStream_t stream) {
  using namespace vit;
  (void)in_sizes; (void)n_in; (void)out_size; (void)ws_size;

  const float* img      = (const float*)d_in[0];
  const int*   mask     = (const int*)d_in[1];
  const float* conv1_w  = (const float*)d_in[2];
  const float* conv1_b  = (const float*)d_in[3];
  const float* ln_p_g   = (const float*)d_in[4];
  const float* ln_p_b   = (const float*)d_in[5];
  const float* proj_w   = (const float*)d_in[6];
  const float* proj_b   = (const float*)d_in[7];
  const float* ln_q_g   = (const float*)d_in[8];
  const float* ln_q_b   = (const float*)d_in[9];
  const float* attn_ln_g= (const float*)d_in[10];
  const float* attn_ln_b= (const float*)d_in[11];
  const float* wqkv     = (const float*)d_in[12];
  const float* wo       = (const float*)d_in[13];
  const float* ff_ln_g  = (const float*)d_in[14];
  const float* ff_ln_b  = (const float*)d_in[15];
  const float* w1       = (const float*)d_in[16];
  const float* b1       = (const float*)d_in[17];
  const float* w2       = (const float*)d_in[18];
  const float* b2       = (const float*)d_in[19];
  const float* conv2_w  = (const float*)d_in[20];
  const float* conv2_b  = (const float*)d_in[21];
  float* out = (float*)d_out;

  // --- workspace carve (bump allocator, 256B aligned) ---
  char* ws = (char*)d_ws;
  size_t off = 0;
  auto alloc = [&](size_t bytes) -> void* {
    void* p = ws + off;
    off += (bytes + 255) & ~(size_t)255;
    return p;
  };
  float*     x      = (float*)alloc((size_t)MROWS * DIM * 4);
  float*     ybuf   = (float*)alloc((size_t)MROWS * DIM * 4);
  float*     ctmp   = (float*)alloc((size_t)BATCH * 128 * IMG * IMG * 4);
  float*     zbuf   = (float*)alloc((size_t)BATCH * 128 * IMG * IMG * 4);
  _Float16*  xn16   = (_Float16*)alloc((size_t)MROWS * DIM * 2);
  float*     qkv    = (float*)alloc((size_t)MROWS * 3 * INNER * 4);
  _Float16*  q16    = (_Float16*)alloc((size_t)BATCH * HEADS * NTOK * DH * 2);
  _Float16*  k16    = (_Float16*)alloc((size_t)BATCH * HEADS * NTOK * DH * 2);
  _Float16*  vT16   = (_Float16*)alloc((size_t)BATCH * HEADS * DH * NTOK * 2);
  _Float16*  o16    = (_Float16*)alloc((size_t)MROWS * INNER * 2);
  _Float16*  hid16  = (_Float16*)alloc((size_t)MROWS * MLP * 2);
  float*     mpart  = (float*)alloc(4096 * 4);
  float*     minv   = (float*)alloc(256);
  int*       rmask  = (int*)alloc((size_t)DEPTH * BATCH * NTOK * 4);
  _Float16*  projT  = (_Float16*)alloc((size_t)DIM * DIM * 2);
  _Float16*  wqkvT  = (_Float16*)alloc((size_t)DEPTH * DIM * 3 * INNER * 2);
  _Float16*  woT    = (_Float16*)alloc((size_t)DEPTH * INNER * DIM * 2);
  _Float16*  w1T    = (_Float16*)alloc((size_t)DEPTH * DIM * MLP * 2);
  _Float16*  w2T    = (_Float16*)alloc((size_t)DEPTH * MLP * DIM * 2);

  auto cdiv = [](size_t a, size_t b) { return (unsigned)((a + b - 1) / b); };

  // --- weight f16 transposes (once per launch; deterministic) ---
  wcast_T<<<cdiv((size_t)DIM * DIM, 256), 256, 0, stream>>>(proj_w, projT, 1, DIM, DIM);
  wcast_T<<<cdiv((size_t)DEPTH * DIM * 3 * INNER, 256), 256, 0, stream>>>(wqkv, wqkvT, DEPTH, DIM, 3 * INNER);
  wcast_T<<<cdiv((size_t)DEPTH * INNER * DIM, 256), 256, 0, stream>>>(wo, woT, DEPTH, INNER, DIM);
  wcast_T<<<cdiv((size_t)DEPTH * DIM * MLP, 256), 256, 0, stream>>>(w1, w1T, DEPTH, DIM, MLP);
  wcast_T<<<cdiv((size_t)DEPTH * MLP * DIM, 256), 256, 0, stream>>>(w2, w2T, DEPTH, MLP, DIM);
  build_rmask<<<cdiv((size_t)DEPTH * BATCH * NTOK, 256), 256, 0, stream>>>(mask, rmask);

  // --- patch embed ---
  conv3x3_k<<<cdiv((size_t)BATCH * 128 * IMG * IMG, 256), 256, 0, stream>>>(
      img, conv1_w, conv1_b, ctmp, BATCH, C_IN, 128);
  patchify<<<cdiv((size_t)BATCH * NTOK * DIM, 256), 256, 0, stream>>>(ctmp, ybuf);
  ln_kernel<false, true><<<MROWS, 256, 0, stream>>>(ybuf, ln_p_g, ln_p_b, nullptr, xn16, DIM);
  gemm_f16_wmma<true, false, false, true, false>
      <<<dim3(DIM / 64, MROWS / 128), 256, 0, stream>>>(
          xn16, projT, proj_b, nullptr, x, nullptr, MROWS, DIM, DIM);
  ln_kernel<true, false><<<MROWS, 256, 0, stream>>>(x, ln_q_g, ln_q_b, x, nullptr, DIM);
  posemb_add<<<cdiv((size_t)BATCH * NTOK * DIM, 256), 256, 0, stream>>>(x);

  // --- transformer layers ---
  const size_t attn_smem = 105984;  // Qs+red+cm+Ss+Ps (dynamic LDS, < 320KB/WGP)
  dim3 agrid(NTOK / 16, HEADS, BATCH);  // 64 x 8 x 8 = 4096 blocks
  for (int i = 0; i < DEPTH; ++i) {
    const int* rmL = rmask + (size_t)i * BATCH * NTOK;
    ln_kernel<false, true><<<MROWS, 256, 0, stream>>>(
        x, attn_ln_g + (size_t)i * DIM, attn_ln_b + (size_t)i * DIM, nullptr, xn16, DIM);
    gemm_f16_wmma<false, false, false, true, false>
        <<<dim3(3 * INNER / 64, MROWS / 128), 256, 0, stream>>>(
            xn16, wqkvT + (size_t)i * DIM * 3 * INNER, nullptr, nullptr,
            qkv, nullptr, MROWS, 3 * INNER, DIM);
    split_qkv<<<cdiv((size_t)BATCH * NTOK * INNER, 256), 256, 0, stream>>>(qkv, q16, k16, vT16);
    attn_wmma<0><<<agrid, 256, attn_smem, stream>>>(q16, k16, vT16, rmL, nullptr, mpart, nullptr);
    reduce_min_k<<<1, 256, 0, stream>>>(mpart, 64 * HEADS * BATCH, minv);
    attn_wmma<1><<<agrid, 256, attn_smem, stream>>>(q16, k16, vT16, rmL, minv, nullptr, o16);
    gemm_f16_wmma<false, true, false, true, false>
        <<<dim3(DIM / 64, MROWS / 128), 256, 0, stream>>>(
            o16, woT + (size_t)i * INNER * DIM, nullptr, x, x, nullptr, MROWS, DIM, INNER);
    ln_kernel<false, true><<<MROWS, 256, 0, stream>>>(
        x, ff_ln_g + (size_t)i * DIM, ff_ln_b + (size_t)i * DIM, nullptr, xn16, DIM);
    gemm_f16_wmma<true, false, true, false, true>
        <<<dim3(MLP / 64, MROWS / 128), 256, 0, stream>>>(
            xn16, w1T + (size_t)i * DIM * MLP, b1 + (size_t)i * MLP, nullptr,
            nullptr, hid16, MROWS, MLP, DIM);
    gemm_f16_wmma<true, true, false, true, false>
        <<<dim3(DIM / 64, MROWS / 128), 256, 0, stream>>>(
            hid16, w2T + (size_t)i * MLP * DIM, b2 + (size_t)i * DIM, x, x,
            nullptr, MROWS, DIM, MLP);
  }

  // --- unpatch + final conv ---
  unpatchify<<<cdiv((size_t)BATCH * 128 * IMG * IMG, 256), 256, 0, stream>>>(x, zbuf);
  conv3x3_k<<<cdiv((size_t)BATCH * C_IN * IMG * IMG, 256), 256, 0, stream>>>(
      zbuf, conv2_w, conv2_b, out, BATCH, 128, C_IN);
}
